// classSR_3class_fused_rcan_net_90168543412496
// MI455X (gfx1250) — matrix operand order repaired and checked by reading, not compiled, not run on hardware
//
#include <hip/hip_runtime.h>
#include <math.h>
#include <stdint.h>

// ---------------------------------------------------------------------------
// ClassSR 3-expert RCAN for MI455X (gfx1250, wave32, WMMA).
// All convs are implicit GEMMs on v_wmma_f32_16x16x32_f16 with channel dims
// zero-padded to 64 (256 for upsample conv outputs). fp16 data, fp32 accum.
// Conv blocks: 4 waves/WGP-block share one LDS halo staged with
// GLOBAL_LOAD_ASYNC_TO_LDS_B128 (ASYNCcnt), each wave owns a 16-ch N-tile.
// ---------------------------------------------------------------------------

#define USE_ASYNC_LDS 1   // flip to 0 if the async mnemonic is rejected

typedef __attribute__((ext_vector_type(16))) _Float16 v16h;
typedef __attribute__((ext_vector_type(8)))  _Float16 v8h;
typedef __attribute__((ext_vector_type(8)))  float    v8f;

#define HB 100        // batch size
#define CP 64         // padded channel count (K granularity = 2 chunks of 32)

// ============================ weight repack ================================
// src: OIHW fp32.  dst fp16 packed [tap(9)][kt(2)][nt(NT)][n(16)][k(32)],
// value = w[co=nt*16+n][cin=kt*32+k][tap], zero outside (O, I).
__global__ void k_repack(const float* __restrict__ w, _Float16* __restrict__ dst,
                         int O, int I, int NT) {
  int total = 9 * 2 * NT * 512;
  for (int i = blockIdx.x * blockDim.x + threadIdx.x; i < total;
       i += gridDim.x * blockDim.x) {
    int k   = i & 31;
    int n   = (i >> 5) & 15;
    int nt  = (i >> 9) % NT;
    int kt  = (i / (512 * NT)) & 1;
    int tap = i / (1024 * NT);
    int cin = kt * 32 + k;
    int co  = nt * 16 + n;
    float v = 0.f;
    if (cin < I && co < O) v = w[(co * I + cin) * 9 + tap];
    dst[i] = (_Float16)v;
  }
}

// ============================ classifier ===================================
__global__ void k_cls_conv1(const float* __restrict__ x, const float* __restrict__ w,
                            const float* __restrict__ b, float* __restrict__ out) {
  int i = blockIdx.x * blockDim.x + threadIdx.x;            // [100][32][32][32]
  if (i >= HB * 32 * 32 * 32) return;
  int px = i & 31, py = (i >> 5) & 31, co = (i >> 10) & 31, s = i >> 15;
  float acc = b[co];
  for (int ci = 0; ci < 3; ci++)
    for (int dy = 0; dy < 3; dy++) {
      int yy = py + dy - 1; if (yy < 0 || yy > 31) continue;
      for (int dx = 0; dx < 3; dx++) {
        int xx = px + dx - 1; if (xx < 0 || xx > 31) continue;
        acc += (x[((s * 3 + ci) * 32 + yy) * 32 + xx] * (1.f / 255.f)) *
               w[((co * 3 + ci) * 3 + dy) * 3 + dx];
      }
    }
  out[i] = fmaxf(acc, 0.f);
}

__global__ void k_cls_conv2(const float* __restrict__ c1, const float* __restrict__ w,
                            const float* __restrict__ b, float* __restrict__ out) {
  int i = blockIdx.x * blockDim.x + threadIdx.x;            // [100][32][16][16]
  if (i >= HB * 32 * 16 * 16) return;
  int px = i & 15, py = (i >> 4) & 15, co = (i >> 8) & 31, s = i >> 13;
  float acc = b[co];
  for (int ci = 0; ci < 32; ci++)
    for (int dy = 0; dy < 3; dy++) {
      int yy = 2 * py + dy; if (yy > 31) continue;          // SAME, stride 2: pad_lo=0
      for (int dx = 0; dx < 3; dx++) {
        int xx = 2 * px + dx; if (xx > 31) continue;
        acc += c1[((s * 32 + ci) * 32 + yy) * 32 + xx] *
               w[((co * 32 + ci) * 3 + dy) * 3 + dx];
      }
    }
  out[i] = fmaxf(acc, 0.f);
}

__global__ void k_cls_head(const float* __restrict__ c2, const float* __restrict__ fcw,
                           const float* __restrict__ fcb,
                           int* __restrict__ idx, float* __restrict__ gate) {
  int s = blockIdx.x, t = threadIdx.x;                      // 32 threads / sample
  __shared__ float g[32];
  float acc = 0.f;
  const float* p = c2 + (s * 32 + t) * 256;
  for (int i = 0; i < 256; i++) acc += p[i];
  g[t] = acc * (1.f / 256.f);
  __syncthreads();
  if (t == 0) {
    float lg[3];
    for (int o = 0; o < 3; o++) {
      float a = fcb[o];
      for (int c = 0; c < 32; c++) a += fcw[o * 32 + c] * g[c];
      lg[o] = a;
    }
    float m = fmaxf(lg[0], fmaxf(lg[1], lg[2]));
    float e0 = expf(lg[0] - m), e1 = expf(lg[1] - m), e2 = expf(lg[2] - m);
    float den = e0 + e1 + e2;
    float p0 = e0 / den, p1 = e1 / den, p2 = e2 / den;
    int best = 0; float bp = p0;
    if (p1 > bp) { bp = p1; best = 1; }
    if (p2 > bp) { bp = p2; best = 2; }
    idx[s] = best; gate[s] = bp;
  }
}

// serial routing: matches jnp.nonzero(size=cap, fill_value=B) order exactly
__global__ void k_route(const int* __restrict__ idx, int* __restrict__ pos,
                        int* __restrict__ countsOut) {
  if (threadIdx.x != 0 || blockIdx.x != 0) return;
  const int caps[3] = {27, 50, 28};
  for (int e = 0; e < 3; e++) {
    int c = 0, tot = 0;
    for (int i = 0; i < HB; i++)
      if (idx[i] == e) { tot++; if (c < caps[e]) pos[e * 64 + c++] = i; }
    for (int j = c; j < 64; j++) pos[e * 64 + j] = HB;
    countsOut[e] = tot < caps[e] ? tot : caps[e];
  }
}

// ===================== dispatch / elementwise helpers ======================
__global__ void k_gather(const float* __restrict__ x, const int* __restrict__ pos,
                         _Float16* __restrict__ dst, int n) {
  int i = blockIdx.x * blockDim.x + threadIdx.x;            // slot*1024 + px
  if (i >= n) return;
  int px = i & 1023, slot = i >> 10;
  int sample = pos[slot];
  __align__(16) _Float16 v[CP];
  #pragma unroll
  for (int c = 0; c < CP; c++) v[c] = (_Float16)0.f;
  if (sample < HB)
    for (int c = 0; c < 3; c++) v[c] = (_Float16)x[(sample * 3 + c) * 1024 + px];
  v8h* d = (v8h*)(dst + (size_t)i * CP);
  #pragma unroll
  for (int j = 0; j < 8; j++) d[j] = *(v8h*)&v[j * 8];
}

__global__ void k_copy8(const v8h* __restrict__ s, v8h* __restrict__ d, int n) {
  int i = blockIdx.x * blockDim.x + threadIdx.x;
  if (i < n) d[i] = s[i];
}

// z += r * att  (channel attention combine; att fp32 per [slot][64])
__global__ void k_combine(_Float16* __restrict__ z, const _Float16* __restrict__ r,
                          const float* __restrict__ att, int n) {
  int i = blockIdx.x * blockDim.x + threadIdx.x;            // (slot*1024+px)*8 + grp
  if (i >= n) return;
  int grp = i & 7, sp = i >> 3, slot = sp >> 10;
  const float* ap = att + slot * 64 + grp * 8;
  size_t base = (size_t)sp * CP + grp * 8;
  #pragma unroll
  for (int j = 0; j < 8; j++) {
    float zv = (float)z[base + j] + (float)r[base + j] * ap[j];
    z[base + j] = (_Float16)zv;
  }
}

__global__ void k_gap(const _Float16* __restrict__ r, float* __restrict__ g, int HW) {
  int slot = blockIdx.x, ch = threadIdx.x;                  // 64 threads
  float acc = 0.f;
  const _Float16* p = r + (size_t)slot * HW * CP + ch;
  for (int i = 0; i < HW; i++) acc += (float)p[(size_t)i * CP];
  g[slot * 64 + ch] = acc / (float)HW;
}

__global__ void k_ca(const float* __restrict__ g, const float* __restrict__ w1,
                     const float* __restrict__ b1, const float* __restrict__ w2,
                     const float* __restrict__ b2, float* __restrict__ att,
                     int nf, int nr) {
  int slot = blockIdx.x, t = threadIdx.x;                   // 64 threads
  __shared__ float a1[8];
  if (t < nr) {
    float a = b1[t];
    const float* gp = g + slot * 64;
    for (int c = 0; c < nf; c++) a += w1[t * nf + c] * gp[c];
    a1[t] = fmaxf(a, 0.f);
  }
  __syncthreads();
  float o = 0.f;
  if (t < nf) {
    float a = b2[t];
    for (int j = 0; j < nr; j++) a += w2[t * nr + j] * a1[j];
    o = 1.f / (1.f + expf(-a));
  }
  att[slot * 64 + t] = o;
}

// ========================= WMMA implicit-GEMM conv =========================
// in : [slot][H*W][64] fp16, SAME 3x3, Cin padded to 64.
// wpk: packed fp16 [tap][kt][nt][16][32].
// 4 waves / block share one LDS halo (async global->LDS); wave w owns N-tile
// nt = blockIdx.z*4 + w.  18 v_wmma per wave.
// mode 0: normal store (+optional elementwise addSrc) to [slot][H*W][64]
// mode 1: fused pixel-shuffle store into next-stage [slot][4HW][64]
// mode 2: tail — gate-weighted scatter into final fp32 output
__global__ void __launch_bounds__(128)
k_conv3x3(const _Float16* __restrict__ in, const _Float16* __restrict__ wpk,
          const float* __restrict__ bias, int biasN,
          const _Float16* __restrict__ addSrc, _Float16* __restrict__ out,
          int H, int W, int NT, int relu, int mode,
          const int* __restrict__ pos, const float* __restrict__ gate,
          float* __restrict__ finalOut) {
  const int tid    = threadIdx.x;
  const int lane   = tid & 31;
  const int wv     = tid >> 5;
  const int tilesX = W >> 4;
  const int tx     = blockIdx.x % tilesX;
  const int y      = blockIdx.x / tilesX;
  const int slot   = blockIdx.y;
  const int nt     = blockIdx.z * 4 + wv;
  const int x0     = tx << 4;

  __shared__ __align__(16) _Float16 smem[3 * 18 * CP];      // 3 rows x 18 cols halo

  const _Float16* inS = in + (size_t)slot * H * W * CP;
  // stage halo: 54 positions x 8 x 16B chunks = 432 chunks across 128 threads
  for (int ck = tid; ck < 432; ck += 128) {
    int p = ck >> 3, j = ck & 7;
    int r = p / 18, c = p % 18;
    int gy = y + r - 1, gx = x0 + c - 1;
    _Float16* d = &smem[(r * 18 + c) * CP + j * 8];
    if (gy >= 0 && gy < H && gx >= 0 && gx < W) {
      const _Float16* s = inS + (size_t)(gy * W + gx) * CP + j * 8;
#if USE_ASYNC_LDS
      unsigned loff = (unsigned)(uintptr_t)d;               // LDS byte offset
      asm volatile("global_load_async_to_lds_b128 %0, %1, off"
                   :: "v"(loff), "v"((unsigned long long)(uintptr_t)s)
                   : "memory");
#else
      *(v8h*)d = *(const v8h*)s;
#endif
    } else {
      v8h zer = {};
      *(v8h*)d = zer;                                       // disjoint addresses
    }
  }
#if USE_ASYNC_LDS
  asm volatile("s_wait_asynccnt 0x0" ::: "memory");
#endif
  __syncthreads();

  if (nt >= NT) return;                                     // idle wave (tail)

  const int m  = lane & 15;   // A: pixel row of fragment / B: output column
  const int gr = lane >> 4;   // half-wave group

  v8f acc = {};
  #pragma unroll
  for (int tap = 0; tap < 9; tap++) {
    const int dy = tap / 3, dx = tap % 3;
    const _Float16* aBase = &smem[(dy * 18 + m + dx) * CP];
    #pragma unroll
    for (int kt = 0; kt < 2; kt++) {
      // A fragment (16x32 f16): group g holds K = g*8..g*8+7 and 16+g*8..16+g*8+7
      const _Float16* ap = aBase + kt * 32 + gr * 8;
      v8h alo = *(const v8h*)ap;
      v8h ahi = *(const v8h*)(ap + 16);
      v16h A;
      #pragma unroll
      for (int j = 0; j < 8; j++) { A[j] = alo[j]; A[j + 8] = ahi[j]; }
      // B fragment (32x16 f16): lane n = m, K = gr*8..gr*8+7 and 16+gr*8..16+gr*8+7
      const _Float16* bp =
          wpk + ((size_t)(tap * 2 + kt) * NT + nt) * 512 + m * 32 + gr * 8;
      v8h blo = *(const v8h*)bp;
      v8h bhi = *(const v8h*)(bp + 16);
      v16h Bm;
      #pragma unroll
      for (int j = 0; j < 8; j++) { Bm[j] = blo[j]; Bm[j + 8] = bhi[j]; }
      acc = __builtin_amdgcn_wmma_f32_16x16x32_f16(false, A, false, Bm,
                                                   (short)0, acc, false, false);
    }
  }

  // C layout: lane owns column N = m; rows M = i + gr*8
  const int chG = nt * 16 + m;
  const float bv = (bias && chG < biasN) ? bias[chG] : 0.f;
  #pragma unroll
  for (int i = 0; i < 8; i++) {
    int M = i + gr * 8;
    float v = acc[i] + bv;
    if (relu) v = fmaxf(v, 0.f);
    int px = x0 + M;
    if (mode == 0) {
      size_t o = ((size_t)slot * H * W + (size_t)y * W + px) * CP + chG;
      if (addSrc) v += (float)addSrc[o];
      out[o] = (_Float16)v;
    } else if (mode == 1) {
      int c2 = chG >> 2, r1 = (chG >> 1) & 1, r2 = chG & 1;
      size_t o = ((size_t)slot * (4 * H * W) +
                  (size_t)(2 * y + r1) * (2 * W) + (2 * px + r2)) * CP + c2;
      out[o] = (_Float16)v;
    } else {
      int sample = pos[slot];
      if (sample < HB && chG < 3) {
        float gv = gate[sample];
        finalOut[(((size_t)sample * 3 + chG) * H + y) * W + px] = gv * v;
      }
    }
  }
}

// ================================ launch ===================================
static inline int cdiv(int a, int b) { return (a + b - 1) / b; }

extern "C" void kernel_launch(void* const* d_in, const int* in_sizes, int n_in,
                              void* d_out, int out_size, void* d_ws, size_t ws_size,
                              hipStream_t stream) {
  (void)in_sizes; (void)n_in; (void)ws_size;
  const int CAPS_[3] = {27, 50, 28};
  const int NF_[3]   = {36, 50, 64};
  const int NR_[3]   = {2, 3, 4};

  // ---- parse flattened inputs (setup_inputs dict order) ----
  int cur = 0;
  auto nx = [&]() { return (const float*)d_in[cur++]; };
  const float* X = nx();
  const float *cls_c1w = nx(), *cls_c1b = nx(), *cls_c2w = nx(), *cls_c2b = nx();
  const float *cls_fcw = nx(), *cls_fcb = nx();
  struct EP {
    const float *head_w, *head_b;
    const float *c1w[2][4], *c1b[2][4], *c2w[2][4], *c2b[2][4];
    const float *caw1[2][4], *cab1[2][4], *caw2[2][4], *cab2[2][4];
    const float *tw[2], *tb[2];
    const float *body_w, *body_b, *up1_w, *up1_b, *up2_w, *up2_b, *tail_w, *tail_b;
  } E[3];
  for (int e = 0; e < 3; e++) {
    E[e].head_w = nx(); E[e].head_b = nx();
    for (int g = 0; g < 2; g++) {
      for (int b = 0; b < 4; b++) {
        E[e].c1w[g][b] = nx(); E[e].c1b[g][b] = nx();
        E[e].c2w[g][b] = nx(); E[e].c2b[g][b] = nx();
        E[e].caw1[g][b] = nx(); E[e].cab1[g][b] = nx();
        E[e].caw2[g][b] = nx(); E[e].cab2[g][b] = nx();
      }
      E[e].tw[g] = nx(); E[e].tb[g] = nx();
    }
    E[e].body_w = nx(); E[e].body_b = nx();
    E[e].up1_w = nx();  E[e].up1_b = nx();
    E[e].up2_w = nx();  E[e].up2_b = nx();
    E[e].tail_w = nx(); E[e].tail_b = nx();
  }

  // ---- workspace carve ----
  size_t off = 0;
  auto alloc = [&](size_t bytes) -> char* {
    off = (off + 255) & ~(size_t)255;
    char* p = (char*)d_ws + off;
    off += bytes;
    return p;
  };
  const size_t A32 = 50ull * 1024 * CP;      // halves, 32x32 stage buffers
  const size_t A64 = 50ull * 4096 * CP;      // after up1+PS
  const size_t A128 = 50ull * 16384 * CP;    // after up2+PS
  _Float16* bX = (_Float16*)alloc(A32 * 2);
  _Float16* bH = (_Float16*)alloc(A32 * 2);
  _Float16* bY = (_Float16*)alloc(A32 * 2);
  _Float16* bZ = (_Float16*)alloc(A32 * 2);
  _Float16* bT = (_Float16*)alloc(A32 * 2);
  _Float16* bR = (_Float16*)alloc(A32 * 2);
  _Float16* bU1 = (_Float16*)alloc(A64 * 2);
  _Float16* bU2 = (_Float16*)alloc(A128 * 2);
  float* cls1 = (float*)bU2;   // alias: classifier runs before experts
  float* cls2 = (float*)bU1;
  float* gapB = (float*)alloc(50 * 64 * 4);
  float* attB = (float*)alloc(50 * 64 * 4);
  float* gateB = (float*)alloc(HB * 4);
  int* idxB = (int*)alloc(HB * 4);
  int* posB = (int*)alloc(3 * 64 * 4);

  const size_t PK64 = 9 * 2 * 4 * 512;       // halves, Cout padded 64
  const size_t PK256 = 9 * 2 * 16 * 512;     // Cout padded 256
  const size_t PK16 = 9 * 2 * 1 * 512;       // Cout padded 16
  _Float16 *wHead[3], *wC1[3][2][4], *wC2[3][2][4], *wTw[3][2];
  _Float16 *wBody[3], *wUp1[3], *wUp2[3], *wTail[3];
  for (int e = 0; e < 3; e++) {
    wHead[e] = (_Float16*)alloc(PK64 * 2);
    for (int g = 0; g < 2; g++)
      for (int b = 0; b < 4; b++) {
        wC1[e][g][b] = (_Float16*)alloc(PK64 * 2);
        wC2[e][g][b] = (_Float16*)alloc(PK64 * 2);
      }
    wTw[e][0] = (_Float16*)alloc(PK64 * 2);
    wTw[e][1] = (_Float16*)alloc(PK64 * 2);
    wBody[e] = (_Float16*)alloc(PK64 * 2);
    wUp1[e] = (_Float16*)alloc(PK256 * 2);
    wUp2[e] = (_Float16*)alloc(PK256 * 2);
    wTail[e] = (_Float16*)alloc(PK16 * 2);
  }

  auto repack = [&](const float* w, _Float16* dst, int O, int I, int NT) {
    int total = 9 * 2 * NT * 512;
    k_repack<<<cdiv(total, 256), 256, 0, stream>>>(w, dst, O, I, NT);
  };
  for (int e = 0; e < 3; e++) {
    int nf = NF_[e];
    repack(E[e].head_w, wHead[e], nf, 3, 4);
    for (int g = 0; g < 2; g++)
      for (int b = 0; b < 4; b++) {
        repack(E[e].c1w[g][b], wC1[e][g][b], nf, nf, 4);
        repack(E[e].c2w[g][b], wC2[e][g][b], nf, nf, 4);
      }
    repack(E[e].tw[0], wTw[e][0], nf, nf, 4);
    repack(E[e].tw[1], wTw[e][1], nf, nf, 4);
    repack(E[e].body_w, wBody[e], nf, nf, 4);
    repack(E[e].up1_w, wUp1[e], 4 * nf, nf, 16);
    repack(E[e].up2_w, wUp2[e], 4 * nf, nf, 16);
    repack(E[e].tail_w, wTail[e], 3, nf, 1);
  }

  // ---- output zero-init (dropped / over-capacity samples stay 0) ----
  hipMemsetAsync(d_out, 0, (size_t)out_size * sizeof(float), stream);

  // ---- classifier + routing ----
  k_cls_conv1<<<cdiv(HB * 32 * 1024, 256), 256, 0, stream>>>(X, cls_c1w, cls_c1b, cls1);
  k_cls_conv2<<<cdiv(HB * 32 * 256, 256), 256, 0, stream>>>(cls1, cls_c2w, cls_c2b, cls2);
  k_cls_head<<<HB, 32, 0, stream>>>(cls2, cls_fcw, cls_fcb, idxB, gateB);
  k_route<<<1, 32, 0, stream>>>(idxB, posB, (int*)d_out + (size_t)HB * 3 * 128 * 128);

  // ---- experts ----
  for (int e = 0; e < 3; e++) {
    const int cap = CAPS_[e], nf = NF_[e], nr = NR_[e];
    const int* pose = posB + e * 64;

    auto conv = [&](const _Float16* in, const _Float16* wpk, const float* bias,
                    int biasN, const _Float16* add, _Float16* outp, int Hs, int Ws,
                    int NT, int relu, int mode) {
      dim3 g((Ws >> 4) * Hs, cap, cdiv(NT, 4));
      k_conv3x3<<<g, 128, 0, stream>>>(in, wpk, bias, biasN, add, outp, Hs, Ws,
                                       NT, relu, mode, pose, gateB, (float*)d_out);
    };

    k_gather<<<cdiv(cap * 1024, 256), 256, 0, stream>>>(X, pose, bX, cap * 1024);
    conv(bX, wHead[e], E[e].head_b, nf, nullptr, bH, 32, 32, 4, 0, 0);
    k_copy8<<<cdiv(cap * 1024 * 8, 256), 256, 0, stream>>>((v8h*)bH, (v8h*)bY,
                                                           cap * 1024 * 8);
    for (int g = 0; g < 2; g++) {
      k_copy8<<<cdiv(cap * 1024 * 8, 256), 256, 0, stream>>>((v8h*)bY, (v8h*)bZ,
                                                             cap * 1024 * 8);
      for (int b = 0; b < 4; b++) {
        conv(bZ, wC1[e][g][b], E[e].c1b[g][b], nf, nullptr, bT, 32, 32, 4, 1, 0);
        conv(bT, wC2[e][g][b], E[e].c2b[g][b], nf, nullptr, bR, 32, 32, 4, 0, 0);
        k_gap<<<cap, 64, 0, stream>>>(bR, gapB, 1024);
        k_ca<<<cap, 64, 0, stream>>>(gapB, E[e].caw1[g][b], E[e].cab1[g][b],
                                     E[e].caw2[g][b], E[e].cab2[g][b], attB, nf, nr);
        k_combine<<<cdiv(cap * 1024 * 8, 256), 256, 0, stream>>>(bZ, bR, attB,
                                                                 cap * 1024 * 8);
      }
      conv(bZ, wTw[e][g], E[e].tb[g], nf, bY, bY, 32, 32, 4, 0, 0);  // group skip
    }
    conv(bY, wBody[e], E[e].body_b, nf, bH, bZ, 32, 32, 4, 0, 0);    // long skip
    conv(bZ, wUp1[e], E[e].up1_b, 4 * nf, nullptr, bU1, 32, 32, 16, 0, 1);  // x2
    conv(bU1, wUp2[e], E[e].up2_b, 4 * nf, nullptr, bU2, 64, 64, 16, 0, 1); // x4
    conv(bU2, wTail[e], E[e].tail_b, 3, nullptr, nullptr, 128, 128, 1, 0, 2);
  }
}
